// Multi_DefectModel_110_22986664968810
// MI455X (gfx1250) — compile-verified, matrix operand order checked by reading, not run on hardware
//
#include <hip/hip_runtime.h>
#include <hip/hip_bf16.h>
#include <math.h>

// ---------------------------------------------------------------------------
// Model dims
// ---------------------------------------------------------------------------
#define BSZ   64
#define MNODE 100
#define NNODE 6400
#define NEDGE 102400
#define NHEAD 4
#define DHEAD 512
#define HD    2048      // NHEAD*DHEAD
#define EMB   768
#define IMG   1024
#define EPSB  1e-5f

// ordered-uint encoding of -inf: bits(-inf)=0xFF800000, sign set -> ~u
#define ENC_NEG_INF 0x007FFFFFu

typedef __attribute__((ext_vector_type(16))) _Float16 v16h;
typedef __attribute__((ext_vector_type(8)))  _Float16 v8h;
typedef __attribute__((ext_vector_type(8)))  float    v8f;

// global (address_space(1)) view of a v8h for guaranteed global_load_b128
typedef __attribute__((address_space(1))) const v8h gc_v8h;

// ---------------------------------------------------------------------------
// Fragment load from a per-lane base pointer with a compile-time-friendly
// element offset: two 16-byte vector loads through explicit AS(1) pointers
// (-> global_load_b128 + imm offset; avoids flat_load & DScnt coupling).
// A-matrix layout (16-bit, 16x32): lanes 0-15 hold rows 0-15 with K=s..s+7 in
// v0..v3 and K=16+s..23+s in v4..v7 (s = 0 for lanes 0-15, 8 for lanes 16-31).
// W is [N,K] row-major (== B^T), so the same layout yields the B fragment.
// ---------------------------------------------------------------------------
__device__ __forceinline__ v16h frag_at(const _Float16* __restrict__ q, int off) {
  v8h lo = *(gc_v8h*)(q + off);
  v8h hi = *(gc_v8h*)(q + off + 16);
  return __builtin_shufflevector(lo, hi, 0, 1, 2, 3, 4, 5, 6, 7,
                                 8, 9, 10, 11, 12, 13, 14, 15);
}

// ---------------------------------------------------------------------------
// C[M, NT*16 cols @ colBase0 + 64*blockIdx.y](ldc) =
//     act( A[M,K] @ W[., K]^T + bias ),  A/W in f16, f32 accumulate.
// Templated on NT (1..4 column tiles per wave): branch-free, fully unrolled.
// K loop unrolled x2 with two named fragment sets (set0/set1) so the software
// double-buffer needs no register copies; per-lane base pointers make all
// fragment loads immediate-offset b128s. kSteps = K/32 must be even (true for
// all K here: 768, 2048, 512, 1024).
// Output: f32 (Cf) or f16 (Ch) -- exactly one is non-null.
// act: 0 = none, 1 = ELU. Requires: M % 16 == 0, K % 64 == 0.
// ---------------------------------------------------------------------------
template <int NT>
__global__ __launch_bounds__(256) void wmma_gemm_f16_t(
    const _Float16* __restrict__ A, const _Float16* __restrict__ W,
    float* __restrict__ Cf, _Float16* __restrict__ Ch,
    const float* __restrict__ bias,
    int M, int K, int colBase0, int ldc, int act) {
  const int lane = threadIdx.x & 31;
  const int rowBase = blockIdx.x * 128 + (threadIdx.x >> 5) * 16;
  if (rowBase >= M) return;                      // wave-uniform exit
  const int colBase = colBase0 + blockIdx.y * 64;

  const int r    = lane & 15;
  const int sOff = (lane >> 4) << 3;             // 0 or 8

  // per-lane base pointers (hoisted): fragment loads become base + imm offset
  const _Float16* qa = A + (size_t)(rowBase + r) * (size_t)K + sOff;
  const _Float16* qw[NT];
#pragma unroll
  for (int t = 0; t < NT; ++t)
    qw[t] = W + (size_t)(colBase + t * 16 + r) * (size_t)K + sOff;

  v8f acc[NT];
#pragma unroll
  for (int t = 0; t < NT; ++t) acc[t] = (v8f){};

  // prologue: K-steps 0 and 1
  v16h a0 = frag_at(qa, 0);
  v16h a1 = frag_at(qa, 32);
  v16h b0[NT], b1[NT];
#pragma unroll
  for (int t = 0; t < NT; ++t) { b0[t] = frag_at(qw[t], 0); b1[t] = frag_at(qw[t], 32); }
  qa += 64;
#pragma unroll
  for (int t = 0; t < NT; ++t) qw[t] += 64;

  for (int k = 64; k < K; k += 64) {
    // consume set0, refill it with K-step k
#pragma unroll
    for (int t = 0; t < NT; ++t)
      acc[t] = __builtin_amdgcn_wmma_f32_16x16x32_f16(false, a0, false, b0[t],
                                                      (short)0, acc[t], false, false);
    __builtin_prefetch(qa + 256, 0, 1);
    a0 = frag_at(qa, 0);
#pragma unroll
    for (int t = 0; t < NT; ++t) b0[t] = frag_at(qw[t], 0);
    // consume set1, refill it with K-step k+32 (covers set0 loads)
#pragma unroll
    for (int t = 0; t < NT; ++t)
      acc[t] = __builtin_amdgcn_wmma_f32_16x16x32_f16(false, a1, false, b1[t],
                                                      (short)0, acc[t], false, false);
    a1 = frag_at(qa, 32);
#pragma unroll
    for (int t = 0; t < NT; ++t) b1[t] = frag_at(qw[t], 32);
    qa += 64;
#pragma unroll
    for (int t = 0; t < NT; ++t) qw[t] += 64;
  }
  // epilogue: last two K-steps
#pragma unroll
  for (int t = 0; t < NT; ++t)
    acc[t] = __builtin_amdgcn_wmma_f32_16x16x32_f16(false, a0, false, b0[t],
                                                    (short)0, acc[t], false, false);
#pragma unroll
  for (int t = 0; t < NT; ++t)
    acc[t] = __builtin_amdgcn_wmma_f32_16x16x32_f16(false, a1, false, b1[t],
                                                    (short)0, acc[t], false, false);

  // C/D layout: VGPR r, lanes 0-15 -> (row r, col lane); lanes 16-31 -> (row r+8)
  const int rOff = rowBase + ((lane >> 4) << 3);
  const int c0   = lane & 15;
#pragma unroll
  for (int t = 0; t < NT; ++t) {
    int col = colBase + t * 16 + c0;
    float bv = bias ? bias[col] : 0.0f;
#pragma unroll
    for (int rr = 0; rr < 8; ++rr) {
      float v = acc[t][rr] + bv;
      if (act == 1) v = v > 0.0f ? v : expm1f(v);
      size_t off = (size_t)(rOff + rr) * (size_t)ldc + col;
      if (Ch) Ch[off] = (_Float16)v;
      else    Cf[off] = v;
    }
  }
}

// ---------------------------------------------------------------------------
// f32 -> f16 conversion (weights once; activations once per GAT hop)
// ---------------------------------------------------------------------------
__global__ void f32_to_f16_k(const float* __restrict__ in,
                             _Float16* __restrict__ out, int n) {
  int t = blockIdx.x * blockDim.x + threadIdx.x;
  if (t < n) out[t] = (_Float16)in[t];
}

// ---------------------------------------------------------------------------
// GAT helpers
// ---------------------------------------------------------------------------
__device__ __forceinline__ unsigned fenc(float f) {
  unsigned u = __float_as_uint(f);
  return (u & 0x80000000u) ? ~u : (u | 0x80000000u);
}
__device__ __forceinline__ float fdec(unsigned k) {
  unsigned u = (k & 0x80000000u) ? (k ^ 0x80000000u) : ~k;
  return __uint_as_float(u);
}

// el/er: one wave per (node, head); dot h[n, head, :] with al/ar
__global__ __launch_bounds__(256) void gat_elr(
    const float* __restrict__ h, const float* __restrict__ al,
    const float* __restrict__ ar, float* __restrict__ el, float* __restrict__ er) {
  int wid  = (blockIdx.x * blockDim.x + threadIdx.x) >> 5;
  int lane = threadIdx.x & 31;
  if (wid >= NNODE * NHEAD) return;
  int n = wid >> 2, hd = wid & 3;
  const float* hp  = h  + (size_t)n * HD + hd * DHEAD;
  const float* alp = al + hd * DHEAD;
  const float* arp = ar + hd * DHEAD;
  float sl = 0.f, sr = 0.f;
  for (int i = lane; i < DHEAD; i += 32) {
    float hv = hp[i]; sl += hv * alp[i]; sr += hv * arp[i];
  }
#pragma unroll
  for (int o = 16; o > 0; o >>= 1) {
    sl += __shfl_down(sl, o, 32);
    sr += __shfl_down(sr, o, 32);
  }
  if (lane == 0) { el[wid] = sl; er[wid] = sr; }
}

// pass A: leaky_relu logits + segment max (ordered-uint atomicMax)
__global__ void gat_edge_logits(const float* __restrict__ el, const float* __restrict__ er,
                                const int* __restrict__ src, const int* __restrict__ dst,
                                float* __restrict__ eW, unsigned* __restrict__ mxU) {
  int t = blockIdx.x * blockDim.x + threadIdx.x;
  if (t >= NEDGE * NHEAD) return;
  int e = t >> 2, hd = t & 3;
  float v = el[src[e] * NHEAD + hd] + er[dst[e] * NHEAD + hd];
  v = v > 0.f ? v : 0.2f * v;
  eW[t] = v;
  atomicMax(&mxU[dst[e] * NHEAD + hd], fenc(v));
}

// pass B: a = exp(e - mx[dst]); segment sum
__global__ void gat_edge_exp(float* __restrict__ eW, const unsigned* __restrict__ mxU,
                             float* __restrict__ sSum, const int* __restrict__ dst) {
  int t = blockIdx.x * blockDim.x + threadIdx.x;
  if (t >= NEDGE * NHEAD) return;
  int e = t >> 2, hd = t & 3;
  float a = __expf(eW[t] - fdec(mxU[dst[e] * NHEAD + hd]));
  eW[t] = a;
  atomicAdd(&sSum[dst[e] * NHEAD + hd], a);
}

// pass C: out[dst] += h[src] * alpha ; one wave per edge, float4 gathers
__global__ __launch_bounds__(256) void gat_edge_aggregate(
    const float* __restrict__ eW, const float* __restrict__ sSum,
    const int* __restrict__ src, const int* __restrict__ dst,
    const float* __restrict__ h, float* __restrict__ out) {
  int wid  = (blockIdx.x * blockDim.x + threadIdx.x) >> 5;
  int lane = threadIdx.x & 31;
  if (wid >= NEDGE) return;
  int s = src[wid], d = dst[wid];
  float alpha[NHEAD];
#pragma unroll
  for (int hd = 0; hd < NHEAD; ++hd)
    alpha[hd] = eW[wid * NHEAD + hd] / sSum[d * NHEAD + hd];
  const float4* hp = (const float4*)(h + (size_t)s * HD);
  float* op = out + (size_t)d * HD;
  for (int i = lane; i < HD / 4; i += 32) {
    float4 hv = hp[i];
    int f = i * 4;
    float al = alpha[f >> 9];
    atomicAdd(&op[f + 0], hv.x * al);
    atomicAdd(&op[f + 1], hv.y * al);
    atomicAdd(&op[f + 2], hv.z * al);
    atomicAdd(&op[f + 3], hv.w * al);
  }
}

// ---------------------------------------------------------------------------
// Elementwise / small kernels
// ---------------------------------------------------------------------------
__global__ void fill_bias_rows(float* __restrict__ out, const float* __restrict__ bias,
                               int rows, int C) {
  int t = blockIdx.x * blockDim.x + threadIdx.x;
  if (t < rows * C) out[t] = bias[t % C];
}
__global__ void fill_u32(unsigned* p, unsigned v, int n) {
  int t = blockIdx.x * blockDim.x + threadIdx.x;
  if (t < n) p[t] = v;
}
__global__ void fill_f32(float* p, float v, int n) {
  int t = blockIdx.x * blockDim.x + threadIdx.x;
  if (t < n) p[t] = v;
}

// BN over columns ([rows, C], per-column stats); f32 in -> f16 out (GEMM input)
__global__ void bn2_apply_h(const float* __restrict__ x, _Float16* __restrict__ y,
                            const float* __restrict__ g, const float* __restrict__ b,
                            const float* __restrict__ m, const float* __restrict__ v,
                            int rows, int C) {
  int t = blockIdx.x * blockDim.x + threadIdx.x;
  if (t >= rows * C) return;
  int c = t % C;
  y[t] = (_Float16)((x[t] - m[c]) * rsqrtf(v[c] + EPSB) * g[c] + b[c]);
}

// BN3 over node-in-graph index (m = n % MNODE); f16 in -> f16 out (GEMM input)
__global__ void bn3_apply_h(const _Float16* __restrict__ x, _Float16* __restrict__ y,
                            const float* __restrict__ g, const float* __restrict__ b,
                            const float* __restrict__ m, const float* __restrict__ v,
                            int C) {
  int t = blockIdx.x * blockDim.x + threadIdx.x;
  if (t >= NNODE * C) return;
  int mi = (t / C) % MNODE;
  y[t] = (_Float16)(((float)x[t] - m[mi]) * rsqrtf(v[mi] + EPSB) * g[mi] + b[mi]);
}

// bbox branch: BN3 on [N,4] then [4->32] FC + ELU (K=4: scalar)
__global__ void pos_branch(const float* __restrict__ pos,
                           const float* __restrict__ g, const float* __restrict__ b,
                           const float* __restrict__ m, const float* __restrict__ v,
                           const float* __restrict__ W, const float* __restrict__ bias,
                           float* __restrict__ out) {
  int t = blockIdx.x * blockDim.x + threadIdx.x;
  if (t >= NNODE * 32) return;
  int n = t >> 5, c = t & 31, mi = n % MNODE;
  float sc = rsqrtf(v[mi] + EPSB) * g[mi];
  float sh = b[mi] - m[mi] * sc;
  float acc = bias[c];
#pragma unroll
  for (int k = 0; k < 4; ++k)
    acc += (pos[n * 4 + k] * sc + sh) * W[c * 4 + k];
  out[t] = acc > 0.f ? acc : expm1f(acc);
}

// h_feature[b, 0:512] = mean over 100 nodes of concat(h480, pos32) -> feats+512
__global__ void mean_concat(const float* __restrict__ h480, const float* __restrict__ pos32,
                            float* __restrict__ feats) {
  int t = blockIdx.x * blockDim.x + threadIdx.x;
  if (t >= BSZ * 512) return;
  int bb = t / 512, c = t % 512;
  float s = 0.f;
  if (c < 480)
    for (int mm = 0; mm < MNODE; ++mm) s += h480[(size_t)(bb * MNODE + mm) * 480 + c];
  else
    for (int mm = 0; mm < MNODE; ++mm) s += pos32[(size_t)(bb * MNODE + mm) * 32 + (c - 480)];
  feats[bb * 1536 + 512 + c] = s * (1.0f / MNODE);
}

// final: BN on feats[64,1536] then [1536->2] linear
__global__ void final_head(const float* __restrict__ feats,
                           const float* __restrict__ g, const float* __restrict__ bb,
                           const float* __restrict__ mm, const float* __restrict__ vv,
                           const float* __restrict__ W, const float* __restrict__ bias,
                           float* __restrict__ out) {
  int t = blockIdx.x * blockDim.x + threadIdx.x;
  if (t >= BSZ * 2) return;
  int row = t >> 1, c = t & 1;
  float acc = bias[c];
  for (int k = 0; k < 1536; ++k) {
    float xv = (feats[row * 1536 + k] - mm[k]) * rsqrtf(vv[k] + EPSB) * g[k] + bb[k];
    acc += xv * W[c * 1536 + k];
  }
  out[t] = acc;
}

// ---------------------------------------------------------------------------
// Host orchestration
// ---------------------------------------------------------------------------
static void launch_gemm_h(const _Float16* A, const _Float16* W,
                          float* Cf, _Float16* Ch, const float* bias,
                          int M, int K, int Nc, int ldc, int act, hipStream_t s) {
  const int gx = (M + 127) / 128;
  const int fullCols = Nc & ~63;
  if (fullCols > 0) {
    dim3 g(gx, fullCols / 64);
    wmma_gemm_f16_t<4><<<g, 256, 0, s>>>(A, W, Cf, Ch, bias, M, K, 0, ldc, act);
  }
  const int rem = (Nc - fullCols) / 16;
  if (rem > 0) {
    dim3 g(gx, 1);
    switch (rem) {
      case 1: wmma_gemm_f16_t<1><<<g, 256, 0, s>>>(A, W, Cf, Ch, bias, M, K, fullCols, ldc, act); break;
      case 2: wmma_gemm_f16_t<2><<<g, 256, 0, s>>>(A, W, Cf, Ch, bias, M, K, fullCols, ldc, act); break;
      default: wmma_gemm_f16_t<3><<<g, 256, 0, s>>>(A, W, Cf, Ch, bias, M, K, fullCols, ldc, act); break;
    }
  }
}

static inline void launch_conv(const float* in, _Float16* out, size_t n, hipStream_t s) {
  f32_to_f16_k<<<((int)n + 255) / 256, 256, 0, s>>>(in, out, (int)n);
}

static void run_gat_attention(const float* h, const float* al, const float* ar,
                              const float* gbias, const int* src, const int* dst,
                              float* el, float* er, unsigned* mxU, float* sSum,
                              float* eW, float* out, hipStream_t s) {
  gat_elr<<<(NNODE * NHEAD * 32 + 255) / 256, 256, 0, s>>>(h, al, ar, el, er);
  // out initialized with bias (covers zero-in-degree nodes exactly)
  fill_bias_rows<<<((size_t)NNODE * HD + 255) / 256, 256, 0, s>>>(out, gbias, NNODE, HD);
  fill_u32<<<(NNODE * NHEAD + 255) / 256, 256, 0, s>>>(mxU, ENC_NEG_INF, NNODE * NHEAD);
  fill_f32<<<(NNODE * NHEAD + 255) / 256, 256, 0, s>>>(sSum, 0.f, NNODE * NHEAD);
  gat_edge_logits<<<(NEDGE * NHEAD + 255) / 256, 256, 0, s>>>(el, er, src, dst, eW, mxU);
  gat_edge_exp<<<(NEDGE * NHEAD + 255) / 256, 256, 0, s>>>(eW, mxU, sSum, dst);
  gat_edge_aggregate<<<(NEDGE * 32 + 255) / 256, 256, 0, s>>>(eW, sSum, src, dst, h, out);
}

extern "C" void kernel_launch(void* const* d_in, const int* in_sizes, int n_in,
                              void* d_out, int out_size, void* d_ws, size_t ws_size,
                              hipStream_t stream) {
  const float* img      = (const float*)d_in[0];
  const float* ftext    = (const float*)d_in[1];
  const float* unix_emb = (const float*)d_in[2];
  const float* pos_emb  = (const float*)d_in[3];
  const int*   src      = (const int*)d_in[4];
  const int*   dst      = (const int*)d_in[5];
  const float* gat1_W   = (const float*)d_in[6];
  const float* gat1_al  = (const float*)d_in[7];
  const float* gat1_ar  = (const float*)d_in[8];
  const float* gat1_b   = (const float*)d_in[9];
  const float* gat2_W   = (const float*)d_in[10];
  const float* gat2_al  = (const float*)d_in[11];
  const float* gat2_ar  = (const float*)d_in[12];
  const float* gat2_b   = (const float*)d_in[13];
  const float* fc_W     = (const float*)d_in[14];
  const float* fc_b     = (const float*)d_in[15];
  const float* hidden_W = (const float*)d_in[16];
  const float* hidden_b = (const float*)d_in[17];
  const float* bnt_g = (const float*)d_in[18], *bnt_b = (const float*)d_in[19];
  const float* bnt_m = (const float*)d_in[20], *bnt_v = (const float*)d_in[21];
  const float* fc_text_W = (const float*)d_in[22], *fc_text_b = (const float*)d_in[23];
  const float* sw_g = (const float*)d_in[24], *sw_b = (const float*)d_in[25];
  const float* sw_m = (const float*)d_in[26], *sw_v = (const float*)d_in[27];
  const float* swinfc_W = (const float*)d_in[28], *swinfc_b = (const float*)d_in[29];
  const float* bg_g = (const float*)d_in[30], *bg_b = (const float*)d_in[31];
  const float* bg_m = (const float*)d_in[32], *bg_v = (const float*)d_in[33];
  const float* fc_gat_W = (const float*)d_in[34], *fc_gat_b = (const float*)d_in[35];
  const float* bb_g = (const float*)d_in[36], *bb_b = (const float*)d_in[37];
  const float* bb_m = (const float*)d_in[38], *bb_v = (const float*)d_in[39];
  const float* fc_bbox_W = (const float*)d_in[40], *fc_bbox_b = (const float*)d_in[41];
  const float* fbn_g = (const float*)d_in[42], *fbn_b = (const float*)d_in[43];
  const float* fbn_m = (const float*)d_in[44], *fbn_v = (const float*)d_in[45];
  const float* final_W = (const float*)d_in[46], *final_b = (const float*)d_in[47];
  float* out = (float*)d_out;
  (void)ws_size; (void)n_in; (void)in_sizes; (void)out_size;

  // ---------------- workspace layout ----------------
  // f32 region: bufA, bufB ([NNODE,HD] each) + attention scratch + feats
  float* ws = (float*)d_ws;
  const size_t NB = (size_t)NNODE * HD;                 // 13,107,200 floats
  float*    bufA  = ws;
  float*    bufB  = ws + NB;
  float*    el    = ws + 2 * NB;
  float*    er    = el + NNODE * NHEAD;
  unsigned* mxU   = (unsigned*)(er + NNODE * NHEAD);
  float*    sSum  = (float*)(mxU + NNODE * NHEAD);
  float*    eW    = sSum + NNODE * NHEAD;
  float*    feats = eW + (size_t)NEDGE * NHEAD;          // [64,1536]
  const size_t F32_TOTAL = 2 * NB + 4 * (size_t)NNODE * NHEAD
                         + (size_t)NEDGE * NHEAD + (size_t)BSZ * 1536;
  // f16 arena (16B-aligned: F32_TOTAL*4 is a multiple of 16)
  _Float16* h16 = (_Float16*)(ws + F32_TOTAL);
  _Float16* actH   = h16;                                // [NNODE,HD]; also unix16
  _Float16* wGat1  = actH  + NB;                         // 2048*768
  _Float16* wGat2  = wGat1 + (size_t)HD * EMB;           // 2048*2048
  _Float16* wFc    = wGat2 + (size_t)HD * HD;            // 512*2048
  _Float16* wHid   = wFc   + (size_t)512 * HD;           // 8*512*512
  _Float16* wFcGat = wHid  + (size_t)8 * 512 * 512;      // 480*512
  _Float16* wSwin  = wFcGat + (size_t)480 * 512;         // 512*1024
  _Float16* wText  = wSwin + (size_t)512 * IMG;          // 512*768
  _Float16* bnimg16 = wText + (size_t)512 * EMB;         // [64,1024]
  _Float16* bntxt16 = bnimg16 + (size_t)BSZ * IMG;       // [64,768]
  // f16 512-wide ping-pong + BN buffer overlay the dead f32 bufA region
  _Float16* h16a   = (_Float16*)bufA;
  _Float16* h16b   = h16a + (size_t)NNODE * 512;
  _Float16* bn512h = h16a + 2 * (size_t)NNODE * 512;
  // f32 graph-head outputs overlay the dead bufB region
  float* h480  = bufB;                                   // [NNODE,480]
  float* pos32 = bufB + (size_t)NNODE * 480;             // [NNODE,32]

  // ---------------- one-time f16 conversions ----------------
  launch_conv(gat1_W,   wGat1,  (size_t)HD * EMB, stream);
  launch_conv(gat2_W,   wGat2,  (size_t)HD * HD, stream);
  launch_conv(fc_W,     wFc,    (size_t)512 * HD, stream);
  launch_conv(hidden_W, wHid,   (size_t)8 * 512 * 512, stream);
  launch_conv(fc_gat_W, wFcGat, (size_t)480 * 512, stream);
  launch_conv(swinfc_W, wSwin,  (size_t)512 * IMG, stream);
  launch_conv(fc_text_W, wText, (size_t)512 * EMB, stream);
  launch_conv(unix_emb, actH,   (size_t)NNODE * EMB, stream);   // unix16 in actH slot

  // --- GAT layer 1 ---
  launch_gemm_h(actH, wGat1, bufA, nullptr, nullptr, NNODE, EMB, HD, HD, 0, stream);
  run_gat_attention(bufA, gat1_al, gat1_ar, gat1_b, src, dst, el, er, mxU, sSum, eW, bufB, stream);
  // --- GAT layer 2 ---
  launch_conv(bufB, actH, NB, stream);
  launch_gemm_h(actH, wGat2, bufA, nullptr, nullptr, NNODE, HD, HD, HD, 0, stream);
  run_gat_attention(bufA, gat2_al, gat2_ar, gat2_b, src, dst, el, er, mxU, sSum, eW, bufB, stream);
  // --- fc (ELU, f16 out) + 8 hidden (ELU, f16 chain) ---
  launch_conv(bufB, actH, NB, stream);
  launch_gemm_h(actH, wFc, nullptr, h16a, fc_b, NNODE, HD, 512, 512, 1, stream);
  _Float16* cur = h16a; _Float16* nxt = h16b;
  for (int i = 0; i < 8; ++i) {
    launch_gemm_h(cur, wHid + (size_t)i * 512 * 512, nullptr, nxt,
                  hidden_b + (size_t)i * 512, NNODE, 512, 512, 512, 1, stream);
    _Float16* tswap = cur; cur = nxt; nxt = tswap;
  }
  // cur == h16a after 8 layers
  // --- graph head: BN3 (f16->f16) -> fc_gat (ELU, f32 out) ---
  bn3_apply_h<<<((size_t)NNODE * 512 + 255) / 256, 256, 0, stream>>>(cur, bn512h, bg_g, bg_b, bg_m, bg_v, 512);
  launch_gemm_h(bn512h, wFcGat, h480, nullptr, fc_gat_b, NNODE, 512, 480, 480, 1, stream);
  // --- bbox branch ---
  pos_branch<<<(NNODE * 32 + 255) / 256, 256, 0, stream>>>(pos_emb, bb_g, bb_b, bb_m, bb_v,
                                                           fc_bbox_W, fc_bbox_b, pos32);
  // --- mean over nodes -> feats[:,512:1024] ---
  mean_concat<<<(BSZ * 512 + 255) / 256, 256, 0, stream>>>(h480, pos32, feats);
  // --- image branch -> feats[:,0:512] ---
  bn2_apply_h<<<(BSZ * IMG + 255) / 256, 256, 0, stream>>>(img, bnimg16, sw_g, sw_b, sw_m, sw_v, BSZ, IMG);
  launch_gemm_h(bnimg16, wSwin, feats + 0, nullptr, swinfc_b, BSZ, IMG, 512, 1536, 1, stream);
  // --- text branch -> feats[:,1024:1536] ---
  bn2_apply_h<<<(BSZ * EMB + 255) / 256, 256, 0, stream>>>(ftext, bntxt16, bnt_g, bnt_b, bnt_m, bnt_v, BSZ, EMB);
  launch_gemm_h(bntxt16, wText, feats + 1024, nullptr, fc_text_b, BSZ, EMB, 512, 1536, 1, stream);
  // --- final BN + linear ---
  final_head<<<1, 128, 0, stream>>>(feats, fbn_g, fbn_b, fbn_m, fbn_v, final_W, final_b, out);
}
